// SphereGlue_36361193128504
// MI455X (gfx1250) — matrix operand order repaired and checked by reading, compiled, not run on hardware
//
#include <hip/hip_runtime.h>

typedef __attribute__((ext_vector_type(16))) __bf16          v16bf;
typedef __attribute__((ext_vector_type(8)))  float           v8f;
typedef __attribute__((ext_vector_type(8)))  unsigned short  ush8;

#define EW(n) dim3((unsigned)(((n)+255)/256)), dim3(256)

__device__ inline unsigned short f2bf(float f) {
  union { float f; unsigned int u; } x; x.f = f;
  unsigned int r = x.u + 0x7FFFu + ((x.u >> 16) & 1u);   // RNE to bf16
  return (unsigned short)(r >> 16);
}

union Frag { v16bf v; ush8 h[2]; unsigned short u[16]; };

// A fragment: two 8-element bf16 runs at +0 and +16 (each 16B, aligned)
__device__ inline void ld_a(const unsigned short* p, Frag& f) {
  const ush8* q = (const ush8*)p;
  f.h[0] = q[0];
  f.h[1] = q[2];
}
// B fragment: two 8-element bf16 runs at +0 and +16 (each 16B, aligned)
__device__ inline void ld_b(const unsigned short* p, Frag& f) {
  const ush8* q = (const ush8*)p;
  f.h[0] = q[0];
  f.h[1] = q[2];
}

// ---------------------------------------------------------------------------
// NT GEMM on pre-converted bf16 operands:
//   C[m,n] = scale * sum_k A[m,k]*B[n,k] (+bias[n]) (+C) (relu) (bf16 store)
// A[z*bsA + m*sAm + k] (bf16), B[z*bsB + n*sBn + k] (bf16),
// C[z*bsC + oC + m*sCm + n*sCn] (f32, or bf16 if flags&4).
// Requirements: M%64==0, N%64==0, K%32==0, sAm%8==0, sBn%8==0, 16B-aligned bases.
// 4 waves/block; each wave computes a 32x32 tile as 2x2 WMMA accumulators.
// ---------------------------------------------------------------------------
__global__ void __launch_bounds__(128)
wmma_gemm_nt(const unsigned short* __restrict__ A, const unsigned short* __restrict__ B,
             float* __restrict__ C, const float* __restrict__ bias,
             int K,
             int sAm, int bsA,
             int sBn, int bsB,
             int sCm, int sCn, int oC, int bsC,
             float scale, int flags)   // flags: 1=relu, 2=accumulate, 4=bf16 output
{
  const int wave = threadIdx.x >> 5;
  const int lane = threadIdx.x & 31;
  const int z    = blockIdx.z;
  const int row0 = blockIdx.x * 64 + (wave & 1) * 32;
  const int col0 = blockIdx.y * 64 + (wave >> 1) * 32;

  const unsigned short* Ab = A + (long)z * bsA;
  const unsigned short* Bb = B + (long)z * bsB;

  const int lh = lane & 15;
  const int hi = lane >> 4;
  const unsigned short* arow0 = Ab + (long)(row0 + lh) * sAm;
  const unsigned short* arow1 = Ab + (long)(row0 + 16 + lh) * sAm;
  const unsigned short* brow0 = Bb + (long)(col0 + lh) * sBn;
  const unsigned short* brow1 = Bb + (long)(col0 + 16 + lh) * sBn;
  const int aoff = hi * 8;    // A halves: K 0-7 / 8-15 (runs repeat at +16)
  const int boff = hi * 8;    // B halves: K 0-7 / 8-15 (runs repeat at +16)

  v8f acc00 = {0.f,0.f,0.f,0.f,0.f,0.f,0.f,0.f};
  v8f acc01 = acc00, acc10 = acc00, acc11 = acc00;

  for (int k0 = 0; k0 < K; k0 += 32) {
    Frag a0f, a1f, b0f, b1f;
    ld_a(arow0 + k0 + aoff, a0f);
    ld_a(arow1 + k0 + aoff, a1f);
    ld_b(brow0 + k0 + boff, b0f);
    ld_b(brow1 + k0 + boff, b1f);
    if (k0 + 32 < K) {  // speculative prefetch of next K slice (global_prefetch_b8)
      __builtin_prefetch((const void*)(arow0 + k0 + 32), 0, 1);
      __builtin_prefetch((const void*)(arow1 + k0 + 32), 0, 1);
      __builtin_prefetch((const void*)(brow0 + k0 + 32), 0, 1);
      __builtin_prefetch((const void*)(brow1 + k0 + 32), 0, 1);
    }
    acc00 = __builtin_amdgcn_wmma_f32_16x16x32_bf16(false, a0f.v, false, b0f.v, (short)0, acc00, false, false);
    acc01 = __builtin_amdgcn_wmma_f32_16x16x32_bf16(false, a0f.v, false, b1f.v, (short)0, acc01, false, false);
    acc10 = __builtin_amdgcn_wmma_f32_16x16x32_bf16(false, a1f.v, false, b0f.v, (short)0, acc10, false, false);
    acc11 = __builtin_amdgcn_wmma_f32_16x16x32_bf16(false, a1f.v, false, b1f.v, (short)0, acc11, false, false);
  }

  const long base = (long)z * bsC + oC;
  const float bc0 = bias ? bias[col0 + lh]      : 0.0f;
  const float bc1 = bias ? bias[col0 + 16 + lh] : 0.0f;

  auto store_tile = [&](v8f acc, int r0, int c0, float bc) {
#pragma unroll
    for (int j = 0; j < 8; ++j) {
      const int m = r0 + hi * 8 + j;     // D layout: lanes 0-15 -> M 0-7, 16-31 -> M 8-15
      const int n = c0 + lh;
      const long ci = base + (long)m * sCm + (long)n * sCn;
      float r = acc[j] * scale + bc;
      if (flags & 4) {
        ((unsigned short*)C)[ci] = f2bf(r);
      } else {
        if (flags & 2) r += C[ci];
        if (flags & 1) r = fmaxf(r, 0.0f);
        C[ci] = r;
      }
    }
  };
  store_tile(acc00, row0,      col0,      bc0);
  store_tile(acc01, row0,      col0 + 16, bc1);
  store_tile(acc10, row0 + 16, col0,      bc0);
  store_tile(acc11, row0 + 16, col0 + 16, bc1);
}

static void gemm(hipStream_t st, const unsigned short* A, const unsigned short* B,
                 float* C, const float* bias,
                 int M, int N, int K,
                 int sAm, int bsA, int sBn, int bsB,
                 int sCm, int sCn, int oC, int bsC,
                 int batch, float scale, int flags)
{
  dim3 g((unsigned)(M / 64), (unsigned)(N / 64), (unsigned)batch);
  wmma_gemm_nt<<<g, dim3(128), 0, st>>>(A, B, C, bias, K, sAm, bsA, sBn, bsB,
                                        sCm, sCn, oC, bsC, scale, flags);
}

// ---------------------------------------------------------------------------
// Elementwise / reduction helpers (conversions fused where data is touched)
// ---------------------------------------------------------------------------
__global__ void k_f2bf(const float* __restrict__ in, unsigned short* __restrict__ out, int n) {
  int i = blockIdx.x * blockDim.x + threadIdx.x;
  if (i < n) out[i] = f2bf(in[i]);
}

__global__ void build_kenc_x(const float* __restrict__ desc, const float* __restrict__ pts,
                             const float* __restrict__ sc, unsigned short* __restrict__ x) {
  int i = blockIdx.x * blockDim.x + threadIdx.x;     // 1024 x 288 (zero-padded 260->288)
  if (i >= 1024 * 288) return;
  int n = i / 288, c = i % 288;
  float v = 0.0f;
  if (c < 256)       v = desc[n * 256 + c];
  else if (c < 259)  v = pts[n * 3 + (c - 256)];
  else if (c == 259) v = sc[n];
  x[i] = f2bf(v);
}

__global__ void pad_w0(const float* __restrict__ w, unsigned short* __restrict__ wp) {
  int i = blockIdx.x * blockDim.x + threadIdx.x;     // 64 x 288 from 64 x 260
  if (i >= 64 * 288) return;
  int r = i / 288, c = i % 288;
  wp[i] = f2bf((c < 260) ? w[r * 260 + c] : 0.0f);
}

__global__ void bn_stats(const float* __restrict__ x, float* __restrict__ stats, int C, int Np) {
  __shared__ float s1[256], s2[256];
  int c = blockIdx.x, t = threadIdx.x;
  float a = 0.f, b = 0.f;
  for (int n = t; n < Np; n += 256) { float v = x[(long)n * C + c]; a += v; b += v * v; }
  s1[t] = a; s2[t] = b; __syncthreads();
  for (int s = 128; s > 0; s >>= 1) {
    if (t < s) { s1[t] += s1[t + s]; s2[t] += s2[t + s]; }
    __syncthreads();
  }
  if (t == 0) { float m = s1[0] / Np; stats[2 * c] = m; stats[2 * c + 1] = s2[0] / Np - m * m; }
}

// BN + ReLU, emitting bf16 for the next GEMM
__global__ void k_bn_apply(const float* __restrict__ x, const float* __restrict__ stats,
                           const float* __restrict__ g, const float* __restrict__ be,
                           unsigned short* __restrict__ out, int C, int total) {
  int i = blockIdx.x * blockDim.x + threadIdx.x;
  if (i >= total) return;
  int c = i % C;
  float m = stats[2 * c], v = stats[2 * c + 1];
  float r = g[c] * (x[i] - m) * rsqrtf(v + 1e-5f) + be[c];
  out[i] = f2bf(fmaxf(r, 0.0f));
}

__global__ void knn_kernel(const float* __restrict__ pts, int* __restrict__ nbr) {
  __shared__ float dist[1024];
  __shared__ float sm[256];
  __shared__ int   si[256];
  int i = blockIdx.x, t = threadIdx.x;
  float px = pts[i * 3], py = pts[i * 3 + 1], pz = pts[i * 3 + 2];
  for (int j = t; j < 1024; j += 256) {
    float dx = px - pts[j * 3], dy = py - pts[j * 3 + 1], dz = pz - pts[j * 3 + 2];
    dist[j] = (j == i) ? 1e30f : (dx * dx + dy * dy + dz * dz);
  }
  __syncthreads();
  for (int kk = 0; kk < 20; ++kk) {
    float best = 1e30f; int bi = 0;
    for (int j = t; j < 1024; j += 256)
      if (dist[j] < best) { best = dist[j]; bi = j; }
    sm[t] = best; si[t] = bi; __syncthreads();
    for (int s = 128; s > 0; s >>= 1) {
      if (t < s && (sm[t + s] < sm[t] || (sm[t + s] == sm[t] && si[t + s] < si[t]))) {
        sm[t] = sm[t + s]; si[t] = si[t + s];
      }
      __syncthreads();
    }
    if (t == 0) { nbr[i * 20 + kk] = si[0]; dist[si[0]] = 1e30f; }
    __syncthreads();
  }
}

__global__ void k_zero(float* p, int n) {
  int i = blockIdx.x * blockDim.x + threadIdx.x;
  if (i < n) p[i] = 0.0f;
}

__global__ void cheb_scatter(const float* __restrict__ t, const int* __restrict__ nbr,
                             float* __restrict__ out) {
  int i = blockIdx.x, c = threadIdx.x;            // 256 channels
  float val = t[i * 256 + c] * (-1.0f / 20.0f);
  for (int kk = 0; kk < 20; ++kk) {
    int d = nbr[i * 20 + kk];
    atomicAdd(&out[d * 256 + c], val);
  }
}

// out_bf = bf16(2a - b)
__global__ void k_comb2_bf(unsigned short* out, const float* a, const float* b, int n) {
  int i = blockIdx.x * blockDim.x + threadIdx.x;
  if (i < n) out[i] = f2bf(2.0f * a[i] - b[i]);
}

__global__ void k_add_inplace(float* a, const float* b, int n) {
  int i = blockIdx.x * blockDim.x + threadIdx.x;
  if (i < n) a[i] += b[i];
}

// first half of concat: bf16(x) -> ycat[:, :256]
__global__ void k_copy_to_cat(const float* __restrict__ x, unsigned short* __restrict__ y, int n) {
  int i = blockIdx.x * blockDim.x + threadIdx.x;
  if (i >= n) return;
  int r = i >> 8, c = i & 255;
  y[r * 512 + c] = f2bf(x[i]);
}

// [n, c=d*4+h] -> [h][n][d]  (head-major, d contiguous), bf16
__global__ void k_split_heads(const float* __restrict__ in, unsigned short* __restrict__ out) {
  int i = blockIdx.x * blockDim.x + threadIdx.x;
  if (i >= 262144) return;
  int n = i >> 8, c = i & 255;
  out[(c & 3) * 65536 + n * 64 + (c >> 2)] = f2bf(in[i]);
}

// [n, c=d*4+h] -> [h][d][n]  (head-major transposed, n contiguous), bf16
__global__ void k_split_heads_T(const float* __restrict__ in, unsigned short* __restrict__ out) {
  int i = blockIdx.x * blockDim.x + threadIdx.x;
  if (i >= 262144) return;
  int n = i >> 8, c = i & 255;
  out[(c & 3) * 65536 + (c >> 2) * 1024 + n] = f2bf(in[i]);
}

// softmax over 1024 cols; reads f32 scores, writes bf16 probabilities
__global__ void k_softmax(const float* __restrict__ sc, unsigned short* __restrict__ pb) {
  __shared__ float sm[256];
  const float* row = sc + (long)blockIdx.x * 1024;
  unsigned short* orow = pb + (long)blockIdx.x * 1024;
  int t = threadIdx.x;
  float mx = -1e30f;
  for (int j = t; j < 1024; j += 256) mx = fmaxf(mx, row[j]);
  sm[t] = mx; __syncthreads();
  for (int s = 128; s > 0; s >>= 1) { if (t < s) sm[t] = fmaxf(sm[t], sm[t + s]); __syncthreads(); }
  mx = sm[0]; __syncthreads();
  float sum = 0.f;
  for (int j = t; j < 1024; j += 256) sum += __expf(row[j] - mx);
  sm[t] = sum; __syncthreads();
  for (int s = 128; s > 0; s >>= 1) { if (t < s) sm[t] += sm[t + s]; __syncthreads(); }
  float inv = 1.0f / sm[0];
  for (int j = t; j < 1024; j += 256) orow[j] = f2bf(__expf(row[j] - mx) * inv);
}

__global__ void fill_bins(float* __restrict__ Z, const float* __restrict__ alpha) {
  int i = blockIdx.x * blockDim.x + threadIdx.x;
  float a = *alpha;
  if (i < 1024)      Z[i * 1025 + 1024] = a;            // last column
  else if (i < 2049) Z[1024 * 1025 + (i - 1024)] = a;   // last row (incl. corner)
}

__global__ void sinkhorn_u(const float* __restrict__ Z, const float* __restrict__ vv,
                           float* __restrict__ u) {
  __shared__ float sm[256];
  int i = blockIdx.x, t = threadIdx.x;
  const float* row = Z + (long)i * 1025;
  float mx = -1e30f;
  for (int j = t; j < 1025; j += 256) mx = fmaxf(mx, row[j] + vv[j]);
  sm[t] = mx; __syncthreads();
  for (int s = 128; s > 0; s >>= 1) { if (t < s) sm[t] = fmaxf(sm[t], sm[t + s]); __syncthreads(); }
  mx = sm[0]; __syncthreads();
  float sum = 0.f;
  for (int j = t; j < 1025; j += 256) sum += __expf(row[j] + vv[j] - mx);
  sm[t] = sum; __syncthreads();
  for (int s = 128; s > 0; s >>= 1) { if (t < s) sm[t] += sm[t + s]; __syncthreads(); }
  if (t == 0) {
    const float NORM = -7.62461899f;                       // -log(2048)
    float lm = (i < 1024) ? NORM : (NORM + 6.93147181f);   // +log(1024)
    u[i] = lm - (logf(sm[0]) + mx);
  }
}

__global__ void sinkhorn_v(const float* __restrict__ Z, const float* __restrict__ uu,
                           float* __restrict__ v) {
  __shared__ float sm[256];
  int j = blockIdx.x, t = threadIdx.x;
  float mx = -1e30f;
  for (int i = t; i < 1025; i += 256) mx = fmaxf(mx, Z[(long)i * 1025 + j] + uu[i]);
  sm[t] = mx; __syncthreads();
  for (int s = 128; s > 0; s >>= 1) { if (t < s) sm[t] = fmaxf(sm[t], sm[t + s]); __syncthreads(); }
  mx = sm[0]; __syncthreads();
  float sum = 0.f;
  for (int i = t; i < 1025; i += 256) sum += __expf(Z[(long)i * 1025 + j] + uu[i] - mx);
  sm[t] = sum; __syncthreads();
  for (int s = 128; s > 0; s >>= 1) { if (t < s) sm[t] += sm[t + s]; __syncthreads(); }
  if (t == 0) {
    const float NORM = -7.62461899f;
    float ln = (j < 1024) ? NORM : (NORM + 6.93147181f);
    v[j] = ln - (logf(sm[0]) + mx);
  }
}

__global__ void final_z(float* __restrict__ Z, const float* __restrict__ u,
                        const float* __restrict__ v) {
  int idx = blockIdx.x * blockDim.x + threadIdx.x;
  if (idx >= 1025 * 1025) return;
  int i = idx / 1025, j = idx % 1025;
  Z[idx] += u[i] + v[j] + 7.62461899f;   // - norm = +log(2048)
}

__global__ void row_argmax(const float* __restrict__ Z, float* __restrict__ mx1,
                           int* __restrict__ idx1) {
  __shared__ float sm[256];
  __shared__ int   si[256];
  int i = blockIdx.x, t = threadIdx.x;
  float best = -1e30f; int bi = 0;
  for (int j = t; j < 1024; j += 256) {
    float v = Z[(long)i * 1025 + j];
    if (v > best) { best = v; bi = j; }
  }
  sm[t] = best; si[t] = bi; __syncthreads();
  for (int s = 128; s > 0; s >>= 1) {
    if (t < s && (sm[t + s] > sm[t] || (sm[t + s] == sm[t] && si[t + s] < si[t]))) {
      sm[t] = sm[t + s]; si[t] = si[t + s];
    }
    __syncthreads();
  }
  if (t == 0) { mx1[i] = sm[0]; idx1[i] = si[0]; }
}

__global__ void col_argmax(const float* __restrict__ Z, int* __restrict__ idx2) {
  __shared__ float sm[256];
  __shared__ int   si[256];
  int j = blockIdx.x, t = threadIdx.x;
  float best = -1e30f; int bi = 0;
  for (int i = t; i < 1024; i += 256) {
    float v = Z[(long)i * 1025 + j];
    if (v > best) { best = v; bi = i; }
  }
  sm[t] = best; si[t] = bi; __syncthreads();
  for (int s = 128; s > 0; s >>= 1) {
    if (t < s && (sm[t + s] > sm[t] || (sm[t + s] == sm[t] && si[t + s] < si[t]))) {
      sm[t] = sm[t + s]; si[t] = si[t + s];
    }
    __syncthreads();
  }
  if (t == 0) idx2[j] = si[0];
}

__global__ void match_kernel(const float* __restrict__ mx1, const int* __restrict__ idx1,
                             const int* __restrict__ idx2, int* __restrict__ matches,
                             float* __restrict__ mscores) {
  int i = blockIdx.x * blockDim.x + threadIdx.x;
  if (i >= 1024) return;
  int j = idx1[i];
  bool mutual = (idx2[j] == i);
  float ms = mutual ? __expf(mx1[i]) : 0.0f;
  mscores[i] = ms;
  matches[i] = (mutual && ms > 0.2f) ? j : -1;
}

__global__ void k_transpose(const float* __restrict__ in, float* __restrict__ out,
                            int N, int C) {   // in [N,C] -> out [C,N]
  int i = blockIdx.x * blockDim.x + threadIdx.x;
  if (i >= N * C) return;
  int n = i / C, c = i % C;
  out[(long)c * N + n] = in[i];
}

// ---------------------------------------------------------------------------
extern "C" void kernel_launch(void* const* d_in, const int* in_sizes, int n_in,
                              void* d_out, int out_size, void* d_ws, size_t ws_size,
                              hipStream_t stream) {
  (void)in_sizes; (void)n_in; (void)out_size; (void)ws_size;

  const float* h1  = (const float*)d_in[0];
  const float* h2  = (const float*)d_in[1];
  const float* uc1 = (const float*)d_in[2];
  const float* uc2 = (const float*)d_in[3];
  const float* sc1 = (const float*)d_in[4];
  const float* sc2 = (const float*)d_in[5];
  const float* kw[5]  = {(const float*)d_in[6], (const float*)d_in[8], (const float*)d_in[10],
                         (const float*)d_in[12], (const float*)d_in[14]};
  const float* kbb[5] = {(const float*)d_in[7], (const float*)d_in[9], (const float*)d_in[11],
                         (const float*)d_in[13], (const float*)d_in[15]};
  const float* kg[4]  = {(const float*)d_in[16], (const float*)d_in[18],
                         (const float*)d_in[20], (const float*)d_in[22]};
  const float* kbe[4] = {(const float*)d_in[17], (const float*)d_in[19],
                         (const float*)d_in[21], (const float*)d_in[23]};
  const float* chw = (const float*)d_in[24];
  const float* chb = (const float*)d_in[25];
  const float* gpw = (const float*)d_in[26];
  const float* gpb = (const float*)d_in[27];
  const float* gmw = (const float*)d_in[28];
  const float* gmb = (const float*)d_in[29];
  const float* g1w = (const float*)d_in[30];
  const float* g1b = (const float*)d_in[31];
  const float* gg  = (const float*)d_in[32];
  const float* gbe = (const float*)d_in[33];
  const float* g2w = (const float*)d_in[34];
  const float* g2b = (const float*)d_in[35];
  const float* fpw = (const float*)d_in[36];
  const float* fpb = (const float*)d_in[37];
  const float* alpha = (const float*)d_in[38];

  float* ws = (float*)d_ws;
  size_t off = 0;
  auto allocf = [&](size_t n) {               // 64B-aligned bump allocator (floats)
    float* p = ws + off;
    off += (n + 15) & ~(size_t)15;
    return p;
  };
  auto allocb = [&](size_t n) {               // bf16 buffers (2 ushorts per float slot)
    return (unsigned short*)allocf((n + 1) / 2);
  };

  // f32 working buffers
  float* a0   = allocf(1024 * 512);
  float* a1   = allocf(1024 * 512);
  float* dk1  = allocf(1024 * 256);
  float* dk2  = allocf(1024 * 256);
  float* c1   = allocf(1024 * 256);
  float* c2   = allocf(1024 * 256);
  int*   nbr  = (int*)allocf(1024 * 20);
  float* t1   = allocf(1024 * 256);
  float* t2   = allocf(1024 * 256);
  float* qb   = allocf(1024 * 256);
  float* kb_  = allocf(1024 * 256);
  float* vb   = allocf(1024 * 256);
  float* sc4  = allocf(4ull * 1024 * 1024);
  float* y1   = allocf(1024 * 512);
  float* stats= allocf(2 * 512);
  float* del0 = allocf(1024 * 256);
  float* del1 = allocf(1024 * 256);
  float* md1  = allocf(1024 * 256);
  float* md2  = allocf(1024 * 256);
  float* uu   = allocf(1025);
  float* vv   = allocf(1025);
  float* mx1  = allocf(1024);
  int*   idx1 = (int*)allocf(1024);
  int*   idx2 = (int*)allocf(1024);

  // bf16 activation buffers
  unsigned short* xbf  = allocb(1024 * 288);
  unsigned short* a0b  = allocb(1024 * 512);
  unsigned short* a1b  = allocb(1024 * 512);
  unsigned short* dkb  = allocb(1024 * 256);
  unsigned short* t1b  = allocb(1024 * 256);
  unsigned short* t2b  = allocb(1024 * 256);
  unsigned short* d1b  = allocb(1024 * 256);
  unsigned short* d2b  = allocb(1024 * 256);
  unsigned short* qhb  = allocb(1024 * 256);
  unsigned short* khb  = allocb(1024 * 256);
  unsigned short* vTb  = allocb(1024 * 256);
  unsigned short* sc4b = allocb(4ull * 1024 * 1024);
  unsigned short* msgb = allocb(1024 * 256);
  unsigned short* ycb  = allocb(1024 * 512);
  unsigned short* y1b  = allocb(1024 * 512);
  unsigned short* md1b = allocb(1024 * 256);
  unsigned short* md2b = allocb(1024 * 256);

  // bf16 weight copies (converted once per call)
  unsigned short* wp0b = allocb(64 * 288);
  unsigned short* kw1b = allocb(128 * 64);
  unsigned short* kw2b = allocb(256 * 128);
  unsigned short* kw3b = allocb(512 * 256);
  unsigned short* kw4b = allocb(256 * 512);
  unsigned short* chwb = allocb(3 * 65536);
  unsigned short* gpwb = allocb(18ull * 3 * 65536);
  unsigned short* gmwb = allocb(18ull * 65536);
  unsigned short* g1wb = allocb(18ull * 262144);
  unsigned short* g2wb = allocb(18ull * 131072);
  unsigned short* fpwb = allocb(65536);

  pad_w0<<<EW(64 * 288), 0, stream>>>(kw[0], wp0b);
  k_f2bf<<<EW(128 * 64),          0, stream>>>(kw[1], kw1b, 128 * 64);
  k_f2bf<<<EW(256 * 128),         0, stream>>>(kw[2], kw2b, 256 * 128);
  k_f2bf<<<EW(512 * 256),         0, stream>>>(kw[3], kw3b, 512 * 256);
  k_f2bf<<<EW(256 * 512),         0, stream>>>(kw[4], kw4b, 256 * 512);
  k_f2bf<<<EW(3 * 65536),         0, stream>>>(chw,  chwb, 3 * 65536);
  k_f2bf<<<EW(18 * 3 * 65536),    0, stream>>>(gpw,  gpwb, 18 * 3 * 65536);
  k_f2bf<<<EW(18 * 65536),        0, stream>>>(gmw,  gmwb, 18 * 65536);
  k_f2bf<<<EW(18 * 262144),       0, stream>>>(g1w,  g1wb, 18 * 262144);
  k_f2bf<<<EW(18 * 131072),       0, stream>>>(g2w,  g2wb, 18 * 131072);
  k_f2bf<<<EW(65536),             0, stream>>>(fpw,  fpwb, 65536);

  const float* descs[2] = {h1, h2};
  const float* ucs[2]   = {uc1, uc2};
  const float* scs[2]   = {sc1, sc2};
  float* ds[2] = {dk1, dk2};
  float* cs[2] = {c1, c2};
  const unsigned short* kwbs[5] = {wp0b, kw1b, kw2b, kw3b, kw4b};
  static const int CH[6] = {288, 64, 128, 256, 512, 256};  // K padded 260->288

  // ---- keypoint encoder + ChebConv for both images ----
  for (int im = 0; im < 2; ++im) {
    build_kenc_x<<<EW(1024 * 288), 0, stream>>>(descs[im], ucs[im], scs[im], xbf);
    const unsigned short* cur = xbf;
    for (int l = 0; l < 4; ++l) {
      float*          out  = (l & 1) ? a1  : a0;
      unsigned short* outb = (l & 1) ? a1b : a0b;
      gemm(stream, cur, kwbs[l], out, kbb[l], 1024, CH[l + 1], CH[l],
           CH[l], 0,  CH[l], 0,  CH[l + 1], 1, 0, 0,  1, 1.0f, 0);
      bn_stats<<<CH[l + 1], 256, 0, stream>>>(out, stats, CH[l + 1], 1024);
      k_bn_apply<<<EW(1024 * CH[l + 1]), 0, stream>>>(out, stats, kg[l], kbe[l],
                                                      outb, CH[l + 1], 1024 * CH[l + 1]);
      cur = outb;
    }
    gemm(stream, cur, kwbs[4], ds[im], kbb[4], 1024, 256, 512,
         512, 0,  512, 0,  256, 1, 0, 0,  1, 1.0f, 0);
    k_f2bf<<<EW(262144), 0, stream>>>(ds[im], dkb, 262144);

    // ChebConv: KNN graph on the unit sphere, K=2 Chebyshev propagation
    knn_kernel<<<1024, 256, 0, stream>>>(ucs[im], nbr);
    k_zero<<<EW(262144), 0, stream>>>(t1, 262144);
    cheb_scatter<<<1024, 256, 0, stream>>>(ds[im], nbr, t1);            // tx1 = prop(tx0)
    k_f2bf<<<EW(262144), 0, stream>>>(t1, t1b, 262144);
    k_zero<<<EW(262144), 0, stream>>>(t2, 262144);
    cheb_scatter<<<1024, 256, 0, stream>>>(t1, nbr, t2);                // prop(tx1)
    k_comb2_bf<<<EW(262144), 0, stream>>>(t2b, t2, ds[im], 262144);     // tx2 = 2*prop(tx1)-tx0
    gemm(stream, dkb, chwb,          cs[im], chb,     1024, 256, 256,
         256, 0,  256, 0,  256, 1, 0, 0,  1, 1.0f, 0);
    gemm(stream, t1b, chwb + 65536,  cs[im], nullptr, 1024, 256, 256,
         256, 0,  256, 0,  256, 1, 0, 0,  1, 1.0f, 2);
    gemm(stream, t2b, chwb + 131072, cs[im], nullptr, 1024, 256, 256,
         256, 0,  256, 0,  256, 1, 0, 0,  1, 1.0f, 2);
  }

  // ---- 18 GNN attention layers (self/cross alternating) ----
  float* D1 = c1;
  float* D2 = c2;
  auto attn = [&](const float* x, const unsigned short* x_bf, const unsigned short* src_bf,
                  int l, float* delta) {
    const unsigned short* pw = gpwb + (size_t)l * 3 * 65536;
    const float*          pb = gpb  + (size_t)l * 3 * 256;
    gemm(stream, x_bf,   pw,          qb,  pb,       1024, 256, 256,
         256, 0,  256, 0,  256, 1, 0, 0,  1, 1.0f, 0);
    gemm(stream, src_bf, pw + 65536,  kb_, pb + 256, 1024, 256, 256,
         256, 0,  256, 0,  256, 1, 0, 0,  1, 1.0f, 0);
    gemm(stream, src_bf, pw + 131072, vb,  pb + 512, 1024, 256, 256,
         256, 0,  256, 0,  256, 1, 0, 0,  1, 1.0f, 0);
    // repack interleaved heads (c = d*4+h) into contiguous per-head bf16 layouts
    k_split_heads  <<<EW(262144), 0, stream>>>(qb,  qhb);   // [h][n][d]
    k_split_heads  <<<EW(262144), 0, stream>>>(kb_, khb);   // [h][m][d]
    k_split_heads_T<<<EW(262144), 0, stream>>>(vb,  vTb);   // [h][d][m]
    // scores[h,n,m] = q_h(n,:) . k_h(m,:) / 8
    gemm(stream, qhb, khb, sc4, nullptr, 1024, 1024, 64,
         64, 65536,  64, 65536,  1024, 1, 0, 1048576,  4, 0.125f, 0);
    k_softmax<<<4096, 256, 0, stream>>>(sc4, sc4b);
    // msg[n, d*4+h] = sum_m prob[h,n,m] * vT[h,d,m]  (bf16 out)
    gemm(stream, sc4b, vTb, (float*)msgb, nullptr, 1024, 64, 1024,
         1024, 1048576,  1024, 65536,  256, 4, 0, 1,  4, 1.0f, 4);
    // y = concat([x, merge(msg)])  (bf16)
    k_copy_to_cat<<<EW(262144), 0, stream>>>(x, ycb, 262144);
    gemm(stream, msgb, gmwb + (size_t)l * 65536, (float*)ycb, gmb + (size_t)l * 256,
         1024, 256, 256,  256, 0,  256, 0,  512, 1, 256, 0,  1, 1.0f, 4);
    // delta = m2( relu(bn( m1(y) )) )
    gemm(stream, ycb, g1wb + (size_t)l * 262144, y1, g1b + (size_t)l * 512,
         1024, 512, 512,  512, 0,  512, 0,  512, 1, 0, 0,  1, 1.0f, 0);
    bn_stats<<<512, 256, 0, stream>>>(y1, stats, 512, 1024);
    k_bn_apply<<<EW(524288), 0, stream>>>(y1, stats, gg + (size_t)l * 512,
                                          gbe + (size_t)l * 512, y1b, 512, 524288);
    gemm(stream, y1b, g2wb + (size_t)l * 131072, delta, g2b + (size_t)l * 256,
         1024, 256, 512,  512, 0,  512, 0,  256, 1, 0, 0,  1, 1.0f, 0);
  };

  for (int l = 0; l < 18; ++l) {
    k_f2bf<<<EW(262144), 0, stream>>>(D1, d1b, 262144);
    k_f2bf<<<EW(262144), 0, stream>>>(D2, d2b, 262144);
    const unsigned short* s0 = (l & 1) ? d2b : d1b;
    const unsigned short* s1 = (l & 1) ? d1b : d2b;
    attn(D1, d1b, s0, l, del0);
    attn(D2, d2b, s1, l, del1);
    k_add_inplace<<<EW(262144), 0, stream>>>(D1, del0, 262144);
    k_add_inplace<<<EW(262144), 0, stream>>>(D2, del1, 262144);
  }

  // ---- final projection + score matrix + Sinkhorn ----
  k_f2bf<<<EW(262144), 0, stream>>>(D1, d1b, 262144);
  k_f2bf<<<EW(262144), 0, stream>>>(D2, d2b, 262144);
  gemm(stream, d1b, fpwb, md1, fpb, 1024, 256, 256,
       256, 0,  256, 0,  256, 1, 0, 0,  1, 1.0f, 0);
  gemm(stream, d2b, fpwb, md2, fpb, 1024, 256, 256,
       256, 0,  256, 0,  256, 1, 0, 0,  1, 1.0f, 0);
  k_f2bf<<<EW(262144), 0, stream>>>(md1, md1b, 262144);
  k_f2bf<<<EW(262144), 0, stream>>>(md2, md2b, 262144);

  float* Z = (float*)d_out;   // 1025 x 1025, top-left 1024x1024 = md1 . md2^T / 16
  gemm(stream, md1b, md2b, Z, nullptr, 1024, 1024, 256,
       256, 0,  256, 0,  1025, 1, 0, 0,  1, 0.0625f, 0);
  fill_bins<<<EW(2049), 0, stream>>>(Z, alpha);

  k_zero<<<EW(2050), 0, stream>>>(uu, 2050);   // vv starts at uu+1040; covers vv[0..1009] only
  for (int it = 0; it < 20; ++it) {
    sinkhorn_u<<<1025, 256, 0, stream>>>(Z, vv, uu);
    sinkhorn_v<<<1025, 256, 0, stream>>>(Z, uu, vv);
  }
  final_z<<<EW(1025 * 1025), 0, stream>>>(Z, uu, vv);

  // ---- mutual-argmax matching + outputs ----
  row_argmax<<<1024, 256, 0, stream>>>(Z, mx1, idx1);
  col_argmax<<<1024, 256, 0, stream>>>(Z, idx2);

  float* out_f    = (float*)d_out;
  int*   matches  = (int*)(out_f + 1050625);
  float* mscores  = out_f + 1051649;
  match_kernel<<<EW(1024), 0, stream>>>(mx1, idx1, idx2, matches, mscores);
  k_transpose<<<EW(262144), 0, stream>>>(md1, out_f + 1052673, 1024, 256);  // (1,256,1024)
  k_transpose<<<EW(262144), 0, stream>>>(md2, out_f + 1314817, 1024, 256);
}
